// MultiHeadCrossAttention_82497731821814
// MI455X (gfx1250) — matrix operand order repaired and checked by reading, not compiled, not run on hardware
//
#include <hip/hip_runtime.h>
#include <hip/hip_bf16.h>
#include <math.h>

// MultiHeadCrossAttention for MI455X (gfx1250, wave32, WMMA).
// Pipeline (all f16 operands, f32 WMMA accumulate):
//   1. cvt x1,x2 -> f16            4. attention (flash-style, 64-key steps, online softmax)
//   2. transpose+cvt Wq,Wkv,Wo     5. out = AO @ WoT + bo  (f32 out)
//   3. Q = x1h@WqT, [K|V] = x2h@WkvT  (K as [b,h,t,d], V transposed [b,h,d,t])

typedef __attribute__((ext_vector_type(16))) _Float16 v16h;
typedef __attribute__((ext_vector_type(8)))  _Float16 v8h;
typedef __attribute__((ext_vector_type(4)))  _Float16 v4h;
typedef __attribute__((ext_vector_type(8)))  float    v8f;
typedef __attribute__((ext_vector_type(4)))  float    v4f;

#define KB  4
#define KT  2048
#define KDM 1024
#define KH  16
#define KDH 64
#define KM  (KB * KT)   // 8192 rows

#define NEG_BIG (-3.0e38f)   // softmax running-max sentinel; expf(NEG_BIG - m) == 0 exactly

union F16Frag { v16h v; v8h h[2]; };

// A-operand (16x32, f16): lane holds row M=lane%16; laneLo: K 0-7,16-23; laneHi: K 8-15,24-31.
static __device__ __forceinline__ v16h load_a_frag(const _Float16* rowp, int k0, int laneHi) {
  F16Frag f;
  const _Float16* p = rowp + k0 + laneHi * 8;
  f.h[0] = *(const v8h*)(p);
  f.h[1] = *(const v8h*)(p + 16);
  return f.v;
}
// B-operand (32x16, f16): lane holds col N=lane%16; laneLo: K 0-7,16-23; laneHi: K 8-15,24-31.
static __device__ __forceinline__ v16h load_b_frag(const _Float16* rowp, int k0, int laneHi) {
  F16Frag f;
  const _Float16* p = rowp + k0 + laneHi * 8;
  f.h[0] = *(const v8h*)(p);
  f.h[1] = *(const v8h*)(p + 16);
  return f.v;
}
static __device__ __forceinline__ v8f wmma16(v16h a, v16h b, v8f c) {
  return __builtin_amdgcn_wmma_f32_16x16x32_f16(false, a, false, b, (short)0, c, false, false);
}
// reduce across each 16-lane half (rows are disjoint between halves in the C layout)
static __device__ __forceinline__ float redmax16(float x) {
#pragma unroll
  for (int m = 1; m < 16; m <<= 1) x = fmaxf(x, __shfl_xor(x, m, 32));
  return x;
}
static __device__ __forceinline__ float redsum16(float x) {
#pragma unroll
  for (int m = 1; m < 16; m <<= 1) x += __shfl_xor(x, m, 32);
  return x;
}

__global__ void __launch_bounds__(256) cvt_f16_kernel(const float* __restrict__ x,
                                                      _Float16* __restrict__ y, int n4) {
  int i = blockIdx.x * blockDim.x + threadIdx.x;
  if (i < n4) {
    v4f v = *((const v4f*)x + i);
    v4h h;
    h.x = (_Float16)v.x; h.y = (_Float16)v.y; h.z = (_Float16)v.z; h.w = (_Float16)v.w;
    *((v4h*)y + i) = h;
  }
}

// Wt[n*Kd + k] = (f16) W[k*Nd + n]
__global__ void __launch_bounds__(256) transpose_cvt_kernel(const float* __restrict__ W,
                                                            _Float16* __restrict__ Wt,
                                                            int Kd, int Nd) {
  int k = blockIdx.x * 32 + (threadIdx.x & 31);
  int n = blockIdx.y * 8 + (threadIdx.x >> 5);
  if (k < Kd && n < Nd) Wt[(size_t)n * Kd + k] = (_Float16)W[(size_t)k * Nd + n];
}

// Wave-tile GEMM: C[32x64] per wave, A [M][1024] f16 row-major, Bt [N][1024] f16 (pre-transposed).
// MODE 0: scatter to Q [b,h,t,d].  MODE 1: scatter to K [b,h,t,d] / V^T [b,h,d,t].
// MODE 2: f32 out = C + bias.
template <int MODE>
__global__ void __launch_bounds__(128)
gemm_kernel(const _Float16* __restrict__ A, const _Float16* __restrict__ Bt,
            _Float16* __restrict__ d0, _Float16* __restrict__ d1,
            const float* __restrict__ bias, float* __restrict__ df) {
  const int lane   = threadIdx.x & 31;
  const int wave   = threadIdx.x >> 5;
  const int lane16 = lane & 15;
  const int laneHi = lane >> 4;
  const int tile   = blockIdx.x * 4 + wave;
  const int mt     = tile & 255;          // M/32 = 256 tiles, fastest -> waves in a block share B
  const int nt64   = tile >> 8;
  const int m0     = mt * 32;
  const int n0     = nt64 * 64;

  const _Float16* arow0 = A + (size_t)(m0 + lane16) * KDM;
  const _Float16* arow1 = arow0 + (size_t)16 * KDM;
  const _Float16* brow  = Bt + (size_t)(n0 + lane16) * KDM;

  v8f acc[2][4] = {};
  for (int k0 = 0; k0 < KDM; k0 += 32) {
    v16h a0 = load_a_frag(arow0, k0, laneHi);
    v16h a1 = load_a_frag(arow1, k0, laneHi);
#pragma unroll
    for (int nt = 0; nt < 4; ++nt) {
      v16h bf = load_b_frag(brow + (size_t)nt * 16 * KDM, k0, laneHi);
      acc[0][nt] = wmma16(a0, bf, acc[0][nt]);
      acc[1][nt] = wmma16(a1, bf, acc[1][nt]);
    }
  }
#pragma unroll
  for (int s = 0; s < 2; ++s) {
#pragma unroll
    for (int nt = 0; nt < 4; ++nt) {
#pragma unroll
      for (int r = 0; r < 8; ++r) {
        const int m = m0 + s * 16 + laneHi * 8 + r;
        const int n = n0 + nt * 16 + lane16;
        const float v = acc[s][nt][r];
        if (MODE == 0) {
          const int b = m >> 11, t = m & 2047, h = n >> 6, d = n & 63;
          d0[(((size_t)b * KH + h) * KT + t) * KDH + d] = (_Float16)v;
        } else if (MODE == 1) {
          const int b = m >> 11, t = m & 2047, h = n >> 7, rr = n & 127;
          if (rr < 64) d0[(((size_t)b * KH + h) * KT + t) * KDH + rr] = (_Float16)v;
          else         d1[(((size_t)b * KH + h) * KDH + (rr - 64)) * KT + t] = (_Float16)v;
        } else {
          df[(size_t)m * KDM + n] = v + bias[n];
        }
      }
    }
  }
}

// Flash attention, 64 keys per step: 8 WMMAs for S = Q K^T (4 tiles x 2 k-steps),
// online softmax with in-register 4-way max/sum before the 16-lane xor tree,
// P staged via LDS (C-layout -> A-layout), 8 WMMAs for O += P V.
__global__ void __launch_bounds__(128)
attn_kernel(const _Float16* __restrict__ Q, const _Float16* __restrict__ Kb,
            const _Float16* __restrict__ Vt, const float* __restrict__ mask,
            _Float16* __restrict__ AO) {
  __shared__ _Float16 pbuf[4][16 * 64];
  const int lane   = threadIdx.x & 31;
  const int wave   = threadIdx.x >> 5;
  const int lane16 = lane & 15;
  const int laneHi = lane >> 4;
  const int tile = blockIdx.x * 4 + wave;  // 0 .. B*H*(T/16)-1
  const int bh = tile >> 7;                // / (T/16 = 128)
  const int qt = tile & 127;
  const int b  = bh >> 4;
  const int h  = bh & 15;

  const _Float16* qbase = Q  + ((size_t)bh * KT + qt * 16 + lane16) * KDH;
  const _Float16* kbase = Kb + (size_t)bh * KT * KDH;
  const _Float16* vroot = Vt + (size_t)bh * KDH * KT;
  const _Float16* vbase = vroot + (size_t)lane16 * KT;
  const float*    mrow  = mask + (size_t)b * KT + lane16;

  const v16h qa0 = load_a_frag(qbase, 0, laneHi);
  const v16h qa1 = load_a_frag(qbase, 32, laneHi);

  v8f o[4] = {};
  float mr[8], lr[8];
#pragma unroll
  for (int r = 0; r < 8; ++r) { mr[r] = NEG_BIG; lr[r] = 0.f; }

  _Float16*       pw = &pbuf[wave][(size_t)laneHi * 8 * 64 + lane16];  // C-layout write
  const _Float16* pr = &pbuf[wave][(size_t)lane16 * 64];               // A-layout read

  for (int kt = 0; kt < KT; kt += 64) {
    // stream next key/value block toward L0/L2 while this block computes
    if (kt + 64 < KT) {
      __builtin_prefetch(kbase + (size_t)(kt + 64 + lane) * KDH, 0, 3);
      __builtin_prefetch(kbase + (size_t)(kt + 96 + lane) * KDH, 0, 3);
      __builtin_prefetch(vroot + (size_t)lane * KT + kt + 64, 0, 3);
      __builtin_prefetch(vroot + (size_t)(32 + lane) * KT + kt + 64, 0, 3);
    }
    // S = Q K^T for 4 key tiles
    v8f s[4];
#pragma unroll
    for (int j = 0; j < 4; ++j) {
      const _Float16* kp = kbase + (size_t)(kt + j * 16 + lane16) * KDH;
      v8f sj = {};
      sj = wmma16(qa0, load_b_frag(kp, 0, laneHi), sj);
      sj = wmma16(qa1, load_b_frag(kp, 32, laneHi), sj);
      s[j] = sj;
    }
    float madd[4];
#pragma unroll
    for (int j = 0; j < 4; ++j)
      madd[j] = (mrow[kt + j * 16] == 0.f) ? -__builtin_inff() : 0.f;

    const float scale = 0.125f;  // 1/sqrt(DH=64)
#pragma unroll
    for (int r = 0; r < 8; ++r) {
      float e0 = s[0][r] * scale + madd[0];
      float e1 = s[1][r] * scale + madd[1];
      float e2 = s[2][r] * scale + madd[2];
      float e3 = s[3][r] * scale + madd[3];
      float mx = redmax16(fmaxf(fmaxf(e0, e1), fmaxf(e2, e3)));
      float mnew  = fmaxf(mr[r], mx);
      float alpha = __expf(mr[r] - mnew);   // first step: exp(NEG_BIG - m) == 0
      float p0 = __expf(e0 - mnew);
      float p1 = __expf(e1 - mnew);
      float p2 = __expf(e2 - mnew);
      float p3 = __expf(e3 - mnew);
      lr[r] = lr[r] * alpha + redsum16((p0 + p1) + (p2 + p3));
      mr[r] = mnew;
      o[0][r] *= alpha; o[1][r] *= alpha; o[2][r] *= alpha; o[3][r] *= alpha;
      pw[r * 64]      = (_Float16)p0;
      pw[r * 64 + 16] = (_Float16)p1;
      pw[r * 64 + 32] = (_Float16)p2;
      pw[r * 64 + 48] = (_Float16)p3;
    }
    asm volatile("s_wait_dscnt 0x0" ::: "memory");
    v16h pa0 = load_a_frag(pr, 0, laneHi);
    v16h pa1 = load_a_frag(pr, 32, laneHi);
#pragma unroll
    for (int nt = 0; nt < 4; ++nt) {
      const _Float16* vp = vbase + (size_t)(nt * 16) * KT + kt;
      o[nt] = wmma16(pa0, load_b_frag(vp, 0, laneHi), o[nt]);
      o[nt] = wmma16(pa1, load_b_frag(vp, 32, laneHi), o[nt]);
    }
  }

  _Float16* outp = AO + ((size_t)b * KT + qt * 16 + laneHi * 8) * KDM + h * KDH + lane16;
#pragma unroll
  for (int r = 0; r < 8; ++r) {
    float inv = 1.0f / lr[r];
#pragma unroll
    for (int nt = 0; nt < 4; ++nt)
      outp[(size_t)r * KDM + nt * 16] = (_Float16)(o[nt][r] * inv);
  }
}

extern "C" void kernel_launch(void* const* d_in, const int* in_sizes, int n_in,
                              void* d_out, int out_size, void* d_ws, size_t ws_size,
                              hipStream_t stream) {
  (void)in_sizes; (void)n_in; (void)out_size; (void)ws_size;
  const float* x1   = (const float*)d_in[0];
  const float* x2   = (const float*)d_in[1];
  const float* mask = (const float*)d_in[2];
  const float* Wq   = (const float*)d_in[3];
  const float* Wkv  = (const float*)d_in[4];
  const float* Wo   = (const float*)d_in[5];
  const float* bo   = (const float*)d_in[6];
  float* out = (float*)d_out;

  char* ws = (char*)d_ws;
  size_t off = 0;
  auto take = [&](size_t bytes) -> void* {
    void* p = ws + off;
    off += (bytes + 255) & ~(size_t)255;
    return p;
  };
  const size_t actBytes = (size_t)KM * KDM * sizeof(_Float16);  // 16 MB
  _Float16* x1h  = (_Float16*)take(actBytes);
  _Float16* x2h  = (_Float16*)take(actBytes);
  _Float16* WqT  = (_Float16*)take((size_t)KDM * KDM * 2);
  _Float16* WkvT = (_Float16*)take((size_t)2 * KDM * KDM * 2);
  _Float16* WoT  = (_Float16*)take((size_t)KDM * KDM * 2);
  _Float16* Qb   = (_Float16*)take(actBytes);
  _Float16* Kbuf = (_Float16*)take(actBytes);
  _Float16* Vt   = (_Float16*)take(actBytes);
  _Float16* AO   = (_Float16*)take(actBytes);

  const int n4 = KM * KDM / 4;
  cvt_f16_kernel<<<(n4 + 255) / 256, 256, 0, stream>>>(x1, x1h, n4);
  cvt_f16_kernel<<<(n4 + 255) / 256, 256, 0, stream>>>(x2, x2h, n4);
  transpose_cvt_kernel<<<dim3(KDM / 32, KDM / 8), 256, 0, stream>>>(Wq, WqT, KDM, KDM);
  transpose_cvt_kernel<<<dim3(KDM / 32, 2 * KDM / 8), 256, 0, stream>>>(Wkv, WkvT, KDM, 2 * KDM);
  transpose_cvt_kernel<<<dim3(KDM / 32, KDM / 8), 256, 0, stream>>>(Wo, WoT, KDM, KDM);

  gemm_kernel<0><<<(256 * 16) / 4, 128, 0, stream>>>(x1h, WqT, Qb, nullptr, nullptr, nullptr);
  gemm_kernel<1><<<(256 * 32) / 4, 128, 0, stream>>>(x2h, WkvT, Kbuf, Vt, nullptr, nullptr);
  attn_kernel<<<(KB * KH * (KT / 16)) / 4, 128, 0, stream>>>(Qb, Kbuf, Vt, mask, AO);
  gemm_kernel<2><<<(256 * 16) / 4, 128, 0, stream>>>(AO, WoT, nullptr, nullptr, bo, out);
}
